// FlowGuidedPCDAlign_29789893165193
// MI455X (gfx1250) — matrix operand order repaired and checked by reading, not compiled, not run on hardware
//
#include <hip/hip_runtime.h>
#include <hip/hip_bf16.h>
#include <math.h>

typedef __attribute__((ext_vector_type(16))) _Float16 v16h;
typedef __attribute__((ext_vector_type(8)))  _Float16 v8h;
typedef __attribute__((ext_vector_type(8)))  float    v8f;

#define BATCH 4
#define DCN_K   576          // G(8) * Cg(8) * KK(9)
#define DCN_NK  18           // 576 / 32
#define DCN_PITCH (DCN_K + 8)

__device__ __forceinline__ int iclamp(int v, int lo, int hi) {
    return v < lo ? lo : (v > hi ? hi : v);
}

// ---------------------------------------------------------------------------
// Weight swizzle: f32 [Cout, Ktot] -> f16 WMMA A-fragments.
// Fragment layout (16-bit A 16x32 per ISA):
//   lane<16 : M = lane,      K = k0 + {0..7, 16..23}
//   lane>=16: M = lane - 16, K = k0 + {8..15, 24..31}
// Output element (t, s, lane, e) at ((t*nk + s)*32 + lane)*16 + e, so each
// lane's 16 halves are 32 contiguous bytes -> global_load_b128 x2 in the GEMM.
// Zero-pads M >= Cout and K >= Ktot (guards removed from hot loops).
// ---------------------------------------------------------------------------
__global__ void swizzle_wgt(const float* __restrict__ w, _Float16* __restrict__ o,
                            int Cout, int Ktot, int nk, int total)
{
    int idx = blockIdx.x * 256 + threadIdx.x;
    if (idx >= total) return;
    int e    = idx & 15;
    int lane = (idx >> 4) & 31;
    int ts   = idx >> 9;
    int s    = ts % nk;
    int t    = ts / nk;
    int M    = t * 16 + (lane & 15);
    int kb   = s * 32 + (lane < 16 ? 0 : 8);
    int K    = (e < 8) ? (kb + e) : (kb + 16 + (e - 8));
    float v  = (M < Cout && K < Ktot) ? w[(size_t)M * Ktot + K] : 0.0f;
    o[idx] = (_Float16)v;
}

// ---------------------------------------------------------------------------
// 3x3 SAME conv over up to 3 channel-concatenated inputs (WMMA f16->f32).
// Block = 128 threads (4 waves). Stage phase: branchless cooperative im2col of
// the full [Kpad x 16] B tile into LDS (transposed [n][k], padded pitch for
// aligned b128 reads). GEMM phase: wave wv computes oc tile blockIdx.y*4+wv;
// hot loop = 2x global b128 (A, preswizzled) + 2x ds b128 (B) + v_wmma.
// ---------------------------------------------------------------------------
__global__ void __launch_bounds__(128)
conv3x3_wmma(const float* __restrict__ in0, int c0,
             const float* __restrict__ in1, int c1, float scale1,
             const float* __restrict__ in2, int c2,
             const _Float16* __restrict__ wsw, const float* __restrict__ bias,
             int Cout, int H, int W, int npix, int nk,
             float* __restrict__ out, int do_lrelu)
{
    extern __shared__ _Float16 Blds[];          // 16 * pitch halves
    const int Cin  = c0 + c1 + c2;
    const int Ktot = Cin * 9;
    const int Kpad = nk * 32;
    const int pitch = Kpad + 8;                 // halves; multiple of 8 -> 16B rows
    const int tid   = threadIdx.y * 32 + threadIdx.x;
    const int tile0 = blockIdx.x * 16;
    const int HW = H * W;
    const int c01 = c0 + c1;

    // ---------------- stage B tile (branchless; loads always in-bounds) ----
    for (int idx = tid; idx < Kpad * 16; idx += 128) {
        int n = idx & 15;
        int k = idx >> 4;
        int kc = (k < Ktot) ? k : 0;
        int ci = kc / 9;
        int kk = kc - ci * 9;
        int ky = kk / 3 - 1;
        int kx = kk - (kk / 3) * 3 - 1;
        int pix = tile0 + n;
        int p   = (pix < npix) ? pix : 0;
        int pb  = p / HW;
        int rem = p - pb * HW;
        int py  = rem / W;
        int px  = rem - py * W;
        int yy = py + ky, xx = px + kx;
        bool inb = (yy >= 0) & (yy < H) & (xx >= 0) & (xx < W) &
                   (k < Ktot) & (pix < npix);
        int yc = iclamp(yy, 0, H - 1);
        int xc = iclamp(xx, 0, W - 1);
        const float* src = (ci < c0) ? in0 : ((ci < c01) ? in1 : in2);
        int   cc  = (ci < c0) ? ci : ((ci < c01) ? ci - c0 : ci - c01);
        int   Cth = (ci < c0) ? c0 : ((ci < c01) ? c1 : c2);
        float sc  = (ci >= c0 && ci < c01) ? scale1 : 1.0f;
        float val = src[(((size_t)pb * Cth + cc) * H + yc) * W + xc];
        Blds[n * pitch + k] = (_Float16)(inb ? val * sc : 0.0f);
    }
    __syncthreads();

    // ---------------- GEMM phase ----------------
    const int lane = threadIdx.x;
    const int t = blockIdx.y * 4 + threadIdx.y;
    const int ntiles = (Cout + 15) >> 4;
    if (t >= ntiles) return;                    // wave-uniform exit
    const int oc0 = t * 16;
    const bool lowhalf = lane < 16;
    const int nB = lane & 15;

    const _Float16* ap = wsw + ((size_t)t * nk) * 512 + lane * 16;
    const _Float16* bp = Blds + nB * pitch + (lowhalf ? 0 : 8);

    v8f acc = {};
    for (int s = 0; s < nk; ++s) {
        v8h a0 = *(const v8h*)(ap);
        v8h a1 = *(const v8h*)(ap + 8);
        v8h b0 = *(const v8h*)(bp);
        v8h b1 = *(const v8h*)(bp + 16);
        ap += 512;
        bp += 32;
        v16h a, b;
#pragma unroll
        for (int e = 0; e < 8; ++e) {
            a[e] = a0[e]; a[e + 8] = a1[e];
            b[e] = b0[e]; b[e + 8] = b1[e];
        }
        acc = __builtin_amdgcn_wmma_f32_16x16x32_f16(false, a, false, b,
                                                     (short)0, acc, false, false);
    }

    // ---------------- epilogue ----------------
    int pix = tile0 + nB;
    if (pix >= npix) return;
    int pb  = pix / HW;
    int rem = pix - pb * HW;
    int py  = rem / W;
    int px  = rem - py * W;
#pragma unroll
    for (int r = 0; r < 8; ++r) {
        int M = oc0 + r + (lowhalf ? 0 : 8);
        if (M < Cout) {
            float v = acc[r] + bias[M];
            if (do_lrelu) v = (v >= 0.0f) ? v : 0.1f * v;
            out[(((size_t)pb * Cout + M) * H + py) * W + px] = v;
        }
    }
}

// ---------------------------------------------------------------------------
// Modulated deformable conv (G=8, Cg=8, 3x3). Stage: 16 pixels x 72 (g,kk)
// points -> modulated bilinear samples for 8 channels each -> LDS B tile
// (transposed [n][k], K = g*72 + c*9 + kk matching einsum('bgckhw,ogck')).
// GEMM: 4 waves x WMMA over K=576; A from preswizzled f16 weights.
// om: [B,216,H,W] = offsets(144, interleaved y,x per (g,kk)) + mask(72).
// flow (optional): flow[1] added to y-offsets, flow[0] to x (flip(1)).
// ---------------------------------------------------------------------------
__global__ void __launch_bounds__(128)
dcn_wmma(const float* __restrict__ x, const float* __restrict__ om,
         const float* __restrict__ flow, const _Float16* __restrict__ wsw,
         const float* __restrict__ bias, int H, int W, int npix,
         float* __restrict__ out, int do_lrelu)
{
    __shared__ _Float16 Blds[16 * DCN_PITCH];
    const int tid   = threadIdx.y * 32 + threadIdx.x;
    const int tile0 = blockIdx.x * 16;
    const int HW = H * W;

    // ---------------- stage phase ----------------
    for (int it = tid; it < 16 * 72; it += 128) {
        int n  = it / 72;
        int gk = it - n * 72;          // g*9 + kk
        int g  = gk / 9;
        int kk = gk - g * 9;
        int pix = tile0 + n;
        int p = (pix < npix) ? pix : 0;
        int b_ = p / HW;
        int rem = p - b_ * HW;
        int yy = rem / W;
        int xx = rem - yy * W;

        size_t base_om = ((size_t)b_ * 216) * HW + (size_t)yy * W + xx;
        float offy = om[base_om + (size_t)(gk * 2 + 0) * HW];
        float offx = om[base_om + (size_t)(gk * 2 + 1) * HW];
        float mval = om[base_om + (size_t)(144 + gk) * HW];
        if (flow) {
            size_t bf = ((size_t)b_ * 2) * HW + (size_t)yy * W + xx;
            offy += flow[bf + HW];     // flow channel 1 -> y
            offx += flow[bf];          // flow channel 0 -> x
        }
        float m  = 1.0f / (1.0f + expf(-mval));
        float sy = offy + (float)(kk / 3 - 1) + (float)yy;
        float sx = offx + (float)(kk - (kk / 3) * 3 - 1) + (float)xx;
        float y0f = floorf(sy), x0f = floorf(sx);
        float wy = sy - y0f, wx = sx - x0f;
        int y0 = (int)y0f, x0 = (int)x0f;
        int y1 = y0 + 1,   x1 = x0 + 1;
        bool vy0 = (y0 >= 0) & (y0 < H), vy1 = (y1 >= 0) & (y1 < H);
        bool vx0 = (x0 >= 0) & (x0 < W), vx1 = (x1 >= 0) & (x1 < W);
        int y0c = iclamp(y0, 0, H - 1), y1c = iclamp(y1, 0, H - 1);
        int x0c = iclamp(x0, 0, W - 1), x1c = iclamp(x1, 0, W - 1);
        float w00 = (vy0 && vx0) ? (1.0f - wy) * (1.0f - wx) : 0.0f;
        float w01 = (vy0 && vx1) ? (1.0f - wy) * wx          : 0.0f;
        float w10 = (vy1 && vx0) ? wy * (1.0f - wx)          : 0.0f;
        float w11 = (vy1 && vx1) ? wy * wx                   : 0.0f;
        int i00 = y0c * W + x0c, i01 = y0c * W + x1c;
        int i10 = y1c * W + x0c, i11 = y1c * W + x1c;
#pragma unroll
        for (int c = 0; c < 8; ++c) {
            const float* xc = x + (((size_t)b_ * 64 + g * 8 + c) * HW);
            float v = w00 * xc[i00] + w01 * xc[i01] + w10 * xc[i10] + w11 * xc[i11];
            Blds[n * DCN_PITCH + (g * 72 + c * 9 + kk)] = (_Float16)(m * v);
        }
    }
    __syncthreads();

    // ---------------- GEMM phase ----------------
    const int lane = threadIdx.x;
    const int t    = threadIdx.y;          // 4 waves cover 64 output channels
    const int oc0  = t * 16;
    const bool lowhalf = lane < 16;
    const int nB = lane & 15;

    const _Float16* ap = wsw + ((size_t)t * DCN_NK) * 512 + lane * 16;
    const _Float16* bp = Blds + nB * DCN_PITCH + (lowhalf ? 0 : 8);

    v8f acc = {};
#pragma unroll
    for (int s = 0; s < DCN_NK; ++s) {
        v8h a0 = *(const v8h*)(ap);
        v8h a1 = *(const v8h*)(ap + 8);
        v8h b0 = *(const v8h*)(bp);
        v8h b1 = *(const v8h*)(bp + 16);
        ap += 512;
        bp += 32;
        v16h a, b;
#pragma unroll
        for (int e = 0; e < 8; ++e) {
            a[e] = a0[e]; a[e + 8] = a1[e];
            b[e] = b0[e]; b[e + 8] = b1[e];
        }
        acc = __builtin_amdgcn_wmma_f32_16x16x32_f16(false, a, false, b,
                                                     (short)0, acc, false, false);
    }

    int pix = tile0 + nB;
    if (pix >= npix) return;
    int b_ = pix / HW;
    int rem = pix - b_ * HW;
    int yy = rem / W;
    int xx = rem - yy * W;
#pragma unroll
    for (int r = 0; r < 8; ++r) {
        int M = oc0 + r + (lowhalf ? 0 : 8);
        float v = acc[r] + bias[M];
        if (do_lrelu) v = (v >= 0.0f) ? v : 0.1f * v;
        out[(((size_t)b_ * 64 + M) * H + yy) * W + xx] = v;
    }
}

// ---------------------------------------------------------------------------
// 2x bilinear upsample, half-pixel centers, edge clamp (align_corners=False).
// ---------------------------------------------------------------------------
__global__ void up2_bilinear(const float* __restrict__ in, float* __restrict__ out,
                             int C, int H, int W, int total)
{
    int idx = blockIdx.x * blockDim.x + threadIdx.x;
    if (idx >= total) return;
    int W2 = 2 * W, H2 = 2 * H;
    int ox = idx % W2;
    int t  = idx / W2;
    int oy = t % H2; t /= H2;
    int c  = t % C;
    int b  = t / C;
    float sy = (oy + 0.5f) * 0.5f - 0.5f;
    float sx = (ox + 0.5f) * 0.5f - 0.5f;
    float y0f = floorf(sy), x0f = floorf(sx);
    float wy = sy - y0f, wx = sx - x0f;
    int y0 = (int)y0f, x0 = (int)x0f;
    int y0c = iclamp(y0, 0, H - 1), y1c = iclamp(y0 + 1, 0, H - 1);
    int x0c = iclamp(x0, 0, W - 1), x1c = iclamp(x0 + 1, 0, W - 1);
    const float* p = in + ((size_t)b * C + c) * H * W;
    float v = (1.0f - wy) * ((1.0f - wx) * p[y0c * W + x0c] + wx * p[y0c * W + x1c])
            +          wy * ((1.0f - wx) * p[y1c * W + x0c] + wx * p[y1c * W + x1c]);
    out[idx] = v;
}

// ---------------------------------------------------------------------------
// Host orchestration
// ---------------------------------------------------------------------------
extern "C" void kernel_launch(void* const* d_in, const int* in_sizes, int n_in,
                              void* d_out, int out_size, void* d_ws, size_t ws_size,
                              hipStream_t stream)
{
    (void)in_sizes; (void)n_in; (void)out_size; (void)ws_size;

    const float* nbr1    = (const float*)d_in[0];
    const float* warped1 = (const float*)d_in[1];
    const float* ref1    = (const float*)d_in[2];
    const float* flow1   = (const float*)d_in[3];
    const float* nbr2    = (const float*)d_in[4];
    const float* warped2 = (const float*)d_in[5];
    const float* ref2    = (const float*)d_in[6];
    const float* flow2   = (const float*)d_in[7];
    const float* nbr3    = (const float*)d_in[8];
    const float* warped3 = (const float*)d_in[9];
    const float* ref3    = (const float*)d_in[10];
    const float* flow3   = (const float*)d_in[11];

    auto Pw = [&](int i) { return (const float*)d_in[12 + 2 * i]; };
    auto Pb = [&](int i) { return (const float*)d_in[13 + 2 * i]; };
    // spec indices:
    // 0 L3_oc1  1 L3_oc2  2 L3_om  3 L3_dcn  4 L2_oc1  5 L2_oc2  6 L2_oc3
    // 7 L2_om   8 L2_dcn  9 L2_fea 10 L1_oc1 11 L1_oc2 12 L1_oc3 13 L1_om
    // 14 L1_dcn 15 L1_fea 16 cas_oc1 17 cas_oc2 18 cas_om 19 cas_dcn
    static const int pCout[20] = {64,64,216,64, 64,64,64,216,64,64,
                                  64,64,64,216,64,64, 64,64,216,64};
    static const int pKtot[20] = {1170,576,576,576, 1170,1152,576,576,576,1152,
                                  1170,1152,576,576,576,1152, 1152,576,576,576};

    float* ws = (float*)d_ws;
    const size_t N1 = (size_t)BATCH * 160 * 160;   // 102400
    const size_t N2 = (size_t)BATCH * 80 * 80;     // 25600
    const size_t N3 = (size_t)BATCH * 40 * 40;     // 6400
    const size_t C1 = 64 * N1, C2 = 64 * N2, C3 = 64 * N3;
    const size_t OMSZ = 216 * N1;

    float* OM = ws;                    // 216-ch offset/mask buffer (largest level)
    float* A  = OM + OMSZ;             // L1-size pool
    float* Bb = A + C1;
    float* Cc = Bb + C1;
    float* D  = Cc + C1;               // L2-size pool (D holds f2 across L1 phase)
    float* E  = D + C2;
    float* F  = E + C2;
    float* G  = F + C2;
    float* P  = G + C2;                // L3-size pool
    float* Q  = P + C3;
    float* R  = Q + C3;
    float* actEnd = R + C3;            // activation pool end (16B aligned)

    // swizzled f16 weights region
    _Float16* wh = (_Float16*)actEnd;
    size_t woff[21];
    woff[0] = 0;
    for (int i = 0; i < 20; ++i) {
        int nt = (pCout[i] + 15) / 16;
        int nk = (pKtot[i] + 31) / 32;
        woff[i + 1] = woff[i] + (size_t)nt * nk * 512;
    }
    for (int i = 0; i < 20; ++i) {
        int nt = (pCout[i] + 15) / 16;
        int nk = (pKtot[i] + 31) / 32;
        int total = nt * nk * 512;
        swizzle_wgt<<<(total + 255) / 256, 256, 0, stream>>>(
            Pw(i), wh + woff[i], pCout[i], pKtot[i], nk, total);
    }

    auto conv = [&](const float* i0, int c0, const float* i1, int c1, float sc1,
                    const float* i2, int c2, int pi, int H, int W,
                    float* o, int lrelu) {
        int Cout = pCout[pi];
        int nk = (pKtot[pi] + 31) / 32;
        int npix = BATCH * H * W;
        dim3 g((npix + 15) / 16, (Cout + 63) / 64);
        dim3 b(32, 4);
        size_t shmem = (size_t)16 * (nk * 32 + 8) * sizeof(_Float16);
        conv3x3_wmma<<<g, b, shmem, stream>>>(i0, c0, i1, c1, sc1, i2, c2,
                                              wh + woff[pi], Pb(pi),
                                              Cout, H, W, npix, nk, o, lrelu);
    };
    auto dcn = [&](const float* xin, const float* omp, const float* fl, int pi,
                   int H, int W, float* o, int lrelu) {
        int npix = BATCH * H * W;
        dim3 g((npix + 15) / 16);
        dim3 b(32, 4);
        dcn_wmma<<<g, b, 0, stream>>>(xin, omp, fl, wh + woff[pi], Pb(pi),
                                      H, W, npix, o, lrelu);
    };
    auto up2 = [&](const float* i, float* o, int H, int W) {
        int total = BATCH * 64 * 4 * H * W;
        up2_bilinear<<<(total + 255) / 256, 256, 0, stream>>>(i, o, 64, H, W, total);
    };

    // ---------------- L3 ----------------
    conv(warped3, 64, ref3, 64, 1.0f, flow3, 2, 0, 40, 40, P, 1);     // o3 = lrelu(L3_oc1)
    conv(P, 64, nullptr, 0, 1.0f, nullptr, 0, 1, 40, 40, Q, 1);       // o3 = lrelu(L3_oc2)
    conv(Q, 64, nullptr, 0, 1.0f, nullptr, 0, 2, 40, 40, OM, 0);      // om3
    dcn(nbr3, OM, flow3, 3, 40, 40, R, 1);                            // f3 = lrelu(fg_dcn)
    up2(Q, G, 40, 40);                                                // o3u

    // ---------------- L2 ----------------
    conv(warped2, 64, ref2, 64, 1.0f, flow2, 2, 4, 80, 80, E, 1);     // o2 = lrelu(L2_oc1)
    conv(E, 64, G, 64, 2.0f, nullptr, 0, 5, 80, 80, F, 1);            // o2 = lrelu(L2_oc2([o2, 2*o3u]))
    conv(F, 64, nullptr, 0, 1.0f, nullptr, 0, 6, 80, 80, E, 1);       // o2 = lrelu(L2_oc3)
    conv(E, 64, nullptr, 0, 1.0f, nullptr, 0, 7, 80, 80, OM, 0);      // om2
    dcn(nbr2, OM, flow2, 8, 80, 80, F, 0);                            // f2 = fg_dcn (no lrelu)
    up2(R, G, 40, 40);                                                // f3u
    conv(F, 64, G, 64, 1.0f, nullptr, 0, 9, 80, 80, D, 1);            // f2 = lrelu(L2_fea([f2, f3u]))
    up2(E, A, 80, 80);                                                // o2u

    // ---------------- L1 ----------------
    conv(warped1, 64, ref1, 64, 1.0f, flow1, 2, 10, 160, 160, Bb, 1); // o1 = lrelu(L1_oc1)
    conv(Bb, 64, A, 64, 2.0f, nullptr, 0, 11, 160, 160, Cc, 1);       // o1 = lrelu(L1_oc2([o1, 2*o2u]))
    conv(Cc, 64, nullptr, 0, 1.0f, nullptr, 0, 12, 160, 160, A, 1);   // o1 = lrelu(L1_oc3)
    conv(A, 64, nullptr, 0, 1.0f, nullptr, 0, 13, 160, 160, OM, 0);   // om1
    dcn(nbr1, OM, flow1, 14, 160, 160, Bb, 0);                        // f1 = fg_dcn (no lrelu)
    up2(D, Cc, 80, 80);                                               // f2u
    conv(Bb, 64, Cc, 64, 1.0f, nullptr, 0, 15, 160, 160, A, 0);       // f1 = L1_fea (no lrelu)

    // ---------------- cascade ----------------
    conv(A, 64, ref1, 64, 1.0f, nullptr, 0, 16, 160, 160, Bb, 1);     // co = lrelu(cas_oc1([f1, ref1]))
    conv(Bb, 64, nullptr, 0, 1.0f, nullptr, 0, 17, 160, 160, Cc, 1);  // co = lrelu(cas_oc2)
    conv(Cc, 64, nullptr, 0, 1.0f, nullptr, 0, 18, 160, 160, OM, 0);  // om_cas
    dcn(A, OM, nullptr, 19, 160, 160, (float*)d_out, 0);              // out = dcn(f1, co)
}
